// CompositionalCodebookLayer2_58394375357112
// MI455X (gfx1250) — matrix operand order repaired and checked by reading, not compiled, not run on hardware
//
#include <hip/hip_runtime.h>
#include <hip/hip_bf16.h>

// Problem constants (match reference)
#define BB   4
#define SS   2048
#define DIM_ 2048
#define CC   8
#define NN   2048
#define DD   256
#define RR   (BB * SS)          // 8192 rows
#define KSTEPS 8                // DD / 32
#define WAVES  8
#define MROWS  16               // rows per wave
#define MBLOCK (WAVES * MROWS)  // 128 rows per workgroup
#define TILEN  32               // codes staged per pipeline step
#define NT     (NN / TILEN)     // 64 steps
#define LDS_ROWB 528            // padded bytes per LDS codebook row (512 + 16)

typedef __attribute__((ext_vector_type(16))) __bf16 v16bf;
typedef __attribute__((ext_vector_type(8)))  __bf16 v8bf;
typedef __attribute__((ext_vector_type(8)))  float  v8f;

// Async DMA: 16 bytes global -> LDS, tracked by ASYNCcnt (no VGPR bounce).
// asm form per CDNA5 ISA: global_load_async_to_lds_b128 vDstLds, vAddr64, off
__device__ __forceinline__ void async_copy16(const void* g, void* l) {
  unsigned long long ga = (unsigned long long)g;
  unsigned int la = (unsigned int)(unsigned long long)l;  // LDS byte offset
  asm volatile("global_load_async_to_lds_b128 %0, %1, off"
               :: "v"(la), "v"(ga) : "memory");
}

__device__ __forceinline__ void wait_async_zero() {
#if __has_builtin(__builtin_amdgcn_s_wait_asynccnt)
  __builtin_amdgcn_s_wait_asynccnt(0);
#else
  asm volatile("s_wait_asynccnt 0x0" ::: "memory");
#endif
}

// Stage TILEN codebook rows (bf16, 512B each) into an LDS buffer.
// 256 threads x 4 chunks of 16B, all via async-to-LDS DMA.
__device__ __forceinline__ void stage_tile(const __bf16* __restrict__ gbase,
                                           unsigned char* lbase, int tid,
                                           bool prefetch_next) {
#pragma unroll
  for (int j = 0; j < 4; ++j) {
    const int ch  = tid + j * 256;  // 0..1023 chunks
    const int r   = ch >> 5;        // row 0..31
    const int col = ch & 31;        // 16B column
    const __bf16* g = gbase + (size_t)r * DD + col * 8;
    unsigned char* l = lbase + r * LDS_ROWB + col * 16;
    async_copy16(g, l);
    if (prefetch_next)  // warm the tile after next (global_prefetch_b8)
      __builtin_prefetch(g + (size_t)TILEN * DD, 0, 3);
  }
}

// ---------------------------------------------------------------------------
// Prep: codebook fp32 -> bf16 (workspace) + c2[n] = sum(bf16(code)^2) in fp32
// ---------------------------------------------------------------------------
__global__ __launch_bounds__(256) void ccl_prep_kernel(
    const float* __restrict__ cb, __bf16* __restrict__ cb16,
    float* __restrict__ c2) {
  const int tid  = threadIdx.x;
  const int lane = tid & 31;
  const int wave = tid >> 5;
  const int row  = blockIdx.x * 8 + wave;  // 0 .. C*N-1

  const float* src = cb + (size_t)row * DD + lane * 8;
  float4 f0 = *(const float4*)(src);
  float4 f1 = *(const float4*)(src + 4);

  v8bf h;
  h[0] = (__bf16)f0.x; h[1] = (__bf16)f0.y; h[2] = (__bf16)f0.z; h[3] = (__bf16)f0.w;
  h[4] = (__bf16)f1.x; h[5] = (__bf16)f1.y; h[6] = (__bf16)f1.z; h[7] = (__bf16)f1.w;
  *(v8bf*)(cb16 + (size_t)row * DD + lane * 8) = h;

  float sq = 0.f;
#pragma unroll
  for (int e = 0; e < 8; ++e) {
    float b = (float)h[e];
    sq = __builtin_fmaf(b, b, sq);
  }
#pragma unroll
  for (int m = 16; m >= 1; m >>= 1) sq += __shfl_xor(sq, m, 32);
  if (lane == 0) c2[row] = sq;
}

// ---------------------------------------------------------------------------
// Main: per (M-block, codebook): bf16 WMMA dots vs all N codes, running
// argmin of c2[n] - 2*dot, then id write + fp32 gather of winning rows.
// Double-buffered async-LDS pipeline, 2 independent WMMA accumulator chains.
// ---------------------------------------------------------------------------
__global__ __launch_bounds__(256) void ccl_snap_kernel(
    const float* __restrict__ x, const float* __restrict__ cb,
    const __bf16* __restrict__ cb16, const float* __restrict__ c2,
    float* __restrict__ out, int* __restrict__ ids) {
  __shared__ __align__(16) unsigned char ldsB[2 * TILEN * LDS_ROWB];

  const int tid  = threadIdx.x;
  const int lane = tid & 31;
  const int wave = tid >> 5;
  const int c    = blockIdx.y;
  const int rowBase = blockIdx.x * MBLOCK + wave * MROWS;
  const int m    = lane & 15;   // row-in-tile (A) / column (B)
  const int half = lane >> 4;   // A: 0 -> K 0..15, 1 -> K 16..31

  // ---- A fragments: 16 rows of x-chunk, bf16, contiguous K layout (k = 16*half + s) ----
  const float* xr = x + (size_t)(rowBase + m) * DIM_ + c * DD;
  const int khalf = half * 16;
  v16bf a[KSTEPS];
#pragma unroll
  for (int ks = 0; ks < KSTEPS; ++ks) {
    const float* p = xr + ks * 32 + khalf;
    float4 f0 = *(const float4*)(p);
    float4 f1 = *(const float4*)(p + 4);
    float4 f2 = *(const float4*)(p + 8);
    float4 f3 = *(const float4*)(p + 12);
    v16bf av;
    av[0]  = (__bf16)f0.x; av[1]  = (__bf16)f0.y; av[2]  = (__bf16)f0.z; av[3]  = (__bf16)f0.w;
    av[4]  = (__bf16)f1.x; av[5]  = (__bf16)f1.y; av[6]  = (__bf16)f1.z; av[7]  = (__bf16)f1.w;
    av[8]  = (__bf16)f2.x; av[9]  = (__bf16)f2.y; av[10] = (__bf16)f2.z; av[11] = (__bf16)f2.w;
    av[12] = (__bf16)f3.x; av[13] = (__bf16)f3.y; av[14] = (__bf16)f3.z; av[15] = (__bf16)f3.w;
    a[ks] = av;
  }

  float minv[8];
  int   mini[8];
#pragma unroll
  for (int v = 0; v < 8; ++v) { minv[v] = 3.4e38f; mini[v] = 0; }

  const __bf16* cbc = cb16 + (size_t)c * NN * DD;
  const float*  c2c = c2 + (size_t)c * NN;

  unsigned char* buf0 = ldsB;
  unsigned char* buf1 = ldsB + TILEN * LDS_ROWB;

  // Prologue: stage tile 0 into buf0.
  stage_tile(cbc, buf0, tid, true);

  for (int nt = 0; nt < NT; ++nt) {
    wait_async_zero();   // tile nt's async DMA (this wave's) has landed
    __syncthreads();     // all waves: tile nt visible; prev compute done
    if (nt + 1 < NT)     // overlap next tile's DMA with this tile's WMMAs
      stage_tile(cbc + (size_t)(nt + 1) * TILEN * DD,
                 ((nt + 1) & 1) ? buf1 : buf0, tid, nt + 2 < NT);

    const unsigned char* cbuf = (nt & 1) ? buf1 : buf0;
    v8f acc0 = {0.f, 0.f, 0.f, 0.f, 0.f, 0.f, 0.f, 0.f};
    v8f acc1 = {0.f, 0.f, 0.f, 0.f, 0.f, 0.f, 0.f, 0.f};
#pragma unroll
    for (int ks = 0; ks < KSTEPS; ++ks) {
      // B fragments: column = code, 16 contiguous K values per lane.
      v16bf b0 = *(const v16bf*)(cbuf + m * LDS_ROWB + ks * 64 + half * 32);
      v16bf b1 = *(const v16bf*)(cbuf + (16 + m) * LDS_ROWB + ks * 64 + half * 32);
      acc0 = __builtin_amdgcn_wmma_f32_16x16x32_bf16(
          false, a[ks], false, b0, (short)0, acc0, false, false);
      acc1 = __builtin_amdgcn_wmma_f32_16x16x32_bf16(
          false, a[ks], false, b1, (short)0, acc1, false, false);
    }

    const int   n0  = nt * TILEN + m;
    const int   n1  = n0 + 16;
    const float c20 = c2c[n0];
    const float c21 = c2c[n1];
#pragma unroll
    for (int v = 0; v < 8; ++v) {
      float s0 = __builtin_fmaf(-2.0f, acc0[v], c20);
      float s1 = __builtin_fmaf(-2.0f, acc1[v], c21);
      bool l0 = (s0 < minv[v]);
      minv[v] = l0 ? s0 : minv[v];
      mini[v] = l0 ? n0 : mini[v];
      bool l1 = (s1 < minv[v]);
      minv[v] = l1 ? s1 : minv[v];
      mini[v] = l1 ? n1 : mini[v];
    }
  }

  // Butterfly argmin across the 16 lanes holding each row's candidates
  // (masks 8..1 stay within a half-wave; row M=v lives in lanes 0-15,
  //  row M=v+8 in lanes 16-31, per the 32-bit C/D layout).
#pragma unroll
  for (int v = 0; v < 8; ++v) {
#pragma unroll
    for (int msk = 8; msk >= 1; msk >>= 1) {
      float ov = __shfl_xor(minv[v], msk, 32);
      int   oi = __shfl_xor(mini[v], msk, 32);
      if (ov < minv[v] || (ov == minv[v] && oi < mini[v])) {
        minv[v] = ov; mini[v] = oi;
      }
    }
  }

  if (m == 0) {
#pragma unroll
    for (int v = 0; v < 8; ++v)
      ids[(size_t)(rowBase + half * 8 + v) * CC + c] = mini[v];
  }

  // Gather winning fp32 code rows: 16 rows x 256 floats, wave-cooperative.
  const float* cbf = cb + (size_t)c * NN * DD;
#pragma unroll
  for (int r = 0; r < 16; ++r) {
    const int srcLane = (r < 8) ? 0 : 16;
    const int id = __shfl(mini[r & 7], srcLane, 32);
    const float4* sp = (const float4*)(cbf + (size_t)id * DD) + lane * 2;
    float4* dp = (float4*)(out + (size_t)(rowBase + r) * DIM_ + c * DD) + lane * 2;
    dp[0] = sp[0];
    dp[1] = sp[1];
  }
}

// ---------------------------------------------------------------------------
extern "C" void kernel_launch(void* const* d_in, const int* in_sizes, int n_in,
                              void* d_out, int out_size, void* d_ws, size_t ws_size,
                              hipStream_t stream) {
  (void)in_sizes; (void)n_in; (void)out_size; (void)ws_size;
  const float* x  = (const float*)d_in[0];
  const float* cb = (const float*)d_in[1];

  __bf16* cb16 = (__bf16*)d_ws;  // 8 MB
  float*  c2   = (float*)((char*)d_ws + (size_t)CC * NN * DD * sizeof(__bf16));

  float* out = (float*)d_out;
  int*   ids = (int*)(out + (size_t)RR * DIM_);

  ccl_prep_kernel<<<CC * NN / 8, 256, 0, stream>>>(cb, cb16, c2);

  dim3 grid(RR / MBLOCK, CC);
  ccl_snap_kernel<<<grid, 256, 0, stream>>>(x, cb, cb16, c2, out, ids);
}
